// EncoderLayer_16295105921360
// MI455X (gfx1250) — compile-verified, matrix operand order checked by reading, not run on hardware
//
#include <hip/hip_runtime.h>
#include <hip/hip_bf16.h>

// =====================================================================
// T5 encoder layer for MI455X (gfx1250, wave32).
// All GEMM-shaped math runs on v_wmma_f32_16x16x32_bf16 (f32 accumulate).
// Workspace requirement: >= 169 MB.
// =====================================================================

typedef __bf16 bf16_t;
typedef __attribute__((ext_vector_type(16))) __bf16        v16bf;
typedef __attribute__((ext_vector_type(8)))  float         v8f;
typedef __attribute__((ext_vector_type(4)))  unsigned int  v4u;

union Frag {
    v16bf v;
    unsigned int u[8];
    v4u q[2];   // q[0] = K-pairs {0,2,4,6}+8hs, q[1] = {16,18,20,22}+8hs
};

// K-pair base index for 16-bit A/B fragments (ISA 7.12.2):
// VGPR v in 0..3 -> K = hs*8 + 2v ; v in 4..7 -> K = 16 + hs*8 + 2(v-4)
__device__ __forceinline__ int frag_kbase(int v, int hs) {
    return (v < 4) ? (v * 2 + hs * 8) : (16 + (v - 4) * 2 + hs * 8);
}

__device__ __forceinline__ v8f wmma_bf16(const Frag& a, const Frag& b, v8f c) {
    return __builtin_amdgcn_wmma_f32_16x16x32_bf16(
        false, a.v, false, b.v, (short)0, c, false, false);
}

__device__ __forceinline__ v8f v8f_zero() {
    v8f z = {0.f, 0.f, 0.f, 0.f, 0.f, 0.f, 0.f, 0.f};
    return z;
}

// ---------------------------------------------------------------------
// fp32 -> bf16 conversion (weights)
// ---------------------------------------------------------------------
__global__ void f32_to_bf16_kernel(const float* __restrict__ in,
                                   bf16_t* __restrict__ out, int n) {
    int i = blockIdx.x * 256 + threadIdx.x;
    if (i < n) out[i] = (bf16_t)in[i];
}

// ---------------------------------------------------------------------
// RMSNorm: one block per row, fp32 in -> bf16 out
// ---------------------------------------------------------------------
__global__ void rmsnorm_kernel(const float* __restrict__ x,
                               const float* __restrict__ scale,
                               bf16_t* __restrict__ out, int D) {
    __shared__ float red[256];
    int row = blockIdx.x;
    const float* xr = x + (size_t)row * D;
    float ss = 0.f;
    for (int i = threadIdx.x; i < D; i += 256) {
        float v = xr[i];
        ss += v * v;
    }
    red[threadIdx.x] = ss;
    __syncthreads();
    for (int s = 128; s > 0; s >>= 1) {
        if (threadIdx.x < s) red[threadIdx.x] += red[threadIdx.x + s];
        __syncthreads();
    }
    float inv = rsqrtf(red[0] / (float)D + 1e-6f);
    for (int i = threadIdx.x; i < D; i += 256)
        out[(size_t)row * D + i] = (bf16_t)(xr[i] * inv * scale[i]);
}

// ---------------------------------------------------------------------
// T5 relative-position bucket (bidirectional, 32 buckets, max_dist 128)
// ---------------------------------------------------------------------
__device__ __forceinline__ int rel_bucket(int rp) {
    int n = -rp;
    int ret = (n < 0) ? 16 : 0;
    n = (n < 0) ? -n : n;
    if (n < 8) return ret + n;
    float f = logf((float)n * 0.125f + 1.1920929e-07f) * (8.0f / 2.7725887222397811f);
    int v = 8 + (int)f;
    v = v < 15 ? v : 15;
    return ret + v;
}

// bias_tab[h][rp + 1023] for rp in [-1023, 1023], pitch 2048
__global__ void bias_table_kernel(const float* __restrict__ rel_emb,
                                  float* __restrict__ tab) {
    int h = blockIdx.x;  // 16 blocks
    for (int j = threadIdx.x; j < 2047; j += 256) {
        int rp = j - 1023;
        tab[h * 2048 + j] = rel_emb[rel_bucket(rp) * 16 + h];
    }
}

// ---------------------------------------------------------------------
// Tiled bf16 GEMM, fp32 accumulate. C[M,N] = A[M,K] x B[K,N].
// Block tile 128x256, 8 waves (2x4), wave tile 64x64, K step 32.
// EPI: 0 = write bf16, 1 = add fp32 residual & write fp32, 2 = ReLU bf16
// ---------------------------------------------------------------------
#define BM 128
#define BN 256
#define BK 32
#define LP 34  // LDS pitch in halves (stride 17 banks -> conflict free)

template <int EPI>
__global__ void gemm_bf16_kernel(const bf16_t* __restrict__ A,
                                 const bf16_t* __restrict__ B,
                                 void* __restrict__ Cout,
                                 const float* __restrict__ Res,
                                 int M, int N, int K) {
    __shared__ bf16_t As[BM * LP];  // [row][k]
    __shared__ bf16_t Bs[BN * LP];  // [n][k]   (transposed)

    int tid = threadIdx.x;
    int wave = tid >> 5, lane = tid & 31;
    int lane16 = lane & 15, hs = lane >> 4;
    int wm = wave >> 2, wn = wave & 3;  // 2 x 4 wave grid, 64x64 per wave
    int m0 = blockIdx.y * BM;
    int n0 = blockIdx.x * BN;

    v8f acc[4][4];
#pragma unroll
    for (int i = 0; i < 4; i++)
#pragma unroll
        for (int j = 0; j < 4; j++) acc[i][j] = v8f_zero();

    const unsigned int* Ag = (const unsigned int*)A;
    const unsigned int* Bg = (const unsigned int*)B;

    for (int k0 = 0; k0 < K; k0 += BK) {
        // A tile: 128 rows x 32 halves = 2048 dwords, coalesced
#pragma unroll
        for (int i = 0; i < 8; i++) {
            int idx = i * 256 + tid;
            int r = idx >> 4;      // row in tile
            int c = idx & 15;      // dword column (2 halves)
            unsigned int d = Ag[((size_t)(m0 + r) * K + k0) / 2 + c];
            *(unsigned int*)&As[r * LP + c * 2] = d;
        }
        // B tile: 32 k-rows x 256 cols, stored transposed [n][k]
#pragma unroll
        for (int i = 0; i < 16; i++) {
            int idx = i * 256 + tid;
            int r = idx >> 7;      // kc 0..31
            int c = idx & 127;     // dword column -> n = 2c
            unsigned int d = Bg[((size_t)(k0 + r) * N + n0) / 2 + c];
            Bs[(c * 2 + 0) * LP + r] = ((const bf16_t*)&d)[0];
            Bs[(c * 2 + 1) * LP + r] = ((const bf16_t*)&d)[1];
        }
        __syncthreads();

        Frag bfv[4];
#pragma unroll
        for (int nt = 0; nt < 4; nt++) {
            int col = wn * 64 + nt * 16 + lane16;
#pragma unroll
            for (int v = 0; v < 8; v++)
                bfv[nt].u[v] = *(const unsigned int*)&Bs[col * LP + frag_kbase(v, hs)];
        }
#pragma unroll
        for (int mt = 0; mt < 4; mt++) {
            Frag af;
            int row = wm * 64 + mt * 16 + lane16;
#pragma unroll
            for (int v = 0; v < 8; v++)
                af.u[v] = *(const unsigned int*)&As[row * LP + frag_kbase(v, hs)];
#pragma unroll
            for (int nt = 0; nt < 4; nt++)
                acc[mt][nt] = wmma_bf16(af, bfv[nt], acc[mt][nt]);
        }
        __syncthreads();
    }

    // Epilogue. C layout: lane holds column n = lane16, rows m = r + 8*hs.
#pragma unroll
    for (int mt = 0; mt < 4; mt++)
#pragma unroll
        for (int nt = 0; nt < 4; nt++)
#pragma unroll
            for (int r = 0; r < 8; r++) {
                int m = m0 + wm * 64 + mt * 16 + r + 8 * hs;
                int n = n0 + wn * 64 + nt * 16 + lane16;
                size_t off = (size_t)m * N + n;
                float vv = acc[mt][nt][r];
                if (EPI == 0) {
                    ((bf16_t*)Cout)[off] = (bf16_t)vv;
                } else if (EPI == 1) {
                    ((float*)Cout)[off] = Res[off] + vv;
                } else {
                    ((bf16_t*)Cout)[off] = (bf16_t)fmaxf(vv, 0.f);
                }
            }
}

// ---------------------------------------------------------------------
// Flash attention (streaming softmax). One wave per (b, h, 16-query tile);
// the 8 waves of a block share the same (b, h) and march over k-blocks in
// lockstep, so the 32x64 V tile is staged (transposed) ONCE per block.
// All inner-loop addresses are loop-carried pointers + immediate offsets.
// ---------------------------------------------------------------------
__global__ void flash_attn_kernel(const bf16_t* __restrict__ Q,
                                  const bf16_t* __restrict__ Kmat,
                                  const bf16_t* __restrict__ V,
                                  const float* __restrict__ bias_tab,
                                  bf16_t* __restrict__ O) {
    const int S = 1024, HD = 64, DM = 1024;
    __shared__ bf16_t Pl[8 * 16 * LP];  // per-wave P tile  (16 x 32)
    __shared__ bf16_t Vs[64 * LP];      // block-shared V tile, transposed [d][kc]
    __shared__ float  BiasS[2048];      // bias row for this (h), rp+1023

    int wave = threadIdx.x >> 5, lane = threadIdx.x & 31;
    int lane16 = lane & 15, hs = lane >> 4;
    int id = blockIdx.x * 8 + wave;      // 0 .. 8191
    int qt = id & 63;
    int h = (id >> 6) & 15;              // identical for all 8 waves in block
    int b = id >> 10;
    int q0 = qt * 16;

    // stage this head's bias row (8KB) into LDS, shared by all waves
    for (int j = threadIdx.x; j < 2047; j += 256)
        BiasS[j] = bias_tab[h * 2048 + j];
    __syncthreads();

    bf16_t* P = &Pl[wave * 16 * LP];

    // Q fragments: each v16bf = two contiguous 16B chunks of the q-row
    Frag qa[2];
    {
        const v4u* qp = (const v4u*)(Q + ((size_t)(b * S + q0 + lane16) * DM + h * HD));
#pragma unroll
        for (int c = 0; c < 2; c++) {
            qa[c].q[0] = qp[c * 4 + hs];
            qa[c].q[1] = qp[c * 4 + hs + 2];
        }
    }

    Frag onesf;
#pragma unroll
    for (int v = 0; v < 8; v++) onesf.u[v] = 0x3F803F80u;  // bf16 1.0 pair

    v8f acc[4];
#pragma unroll
    for (int i = 0; i < 4; i++) acc[i] = v8f_zero();
    v8f accl = v8f_zero();   // running softmax denominator (ones-column)
    float mrow[8];
#pragma unroll
    for (int r = 0; r < 8; r++) mrow[r] = -1e30f;

    // ---- loop-carried pointers (advance by 32 rows per k-block) ----
    // K: one base pointer; t=0/1 tiles and both d-chunks are constant
    // v4u indices (16*DM halves = 2048 v4u -> folded into ioffset).
    const v4u* kp = (const v4u*)(Kmat + ((size_t)(b * S + lane16) * DM + h * HD));
    // V cooperative stage: chunk i covers kc = wave + 8i, dword column = lane
    const bf16_t* vrow = V + ((size_t)(b * S + wave) * DM + h * HD) + 2 * lane;
    // LDS destination: Vs[(2*lane + {0,1})*LP + wave + 8i]
    bf16_t* vsb = &Vs[(2 * lane) * LP + wave];
    // bias row pointer, indexed by {t*16 - r} then advanced by 32
    const float* biasp = &BiasS[lane16 - 8 * hs - q0 + 1023];

    const int KSTEP = 32 * DM;          // halves per k-block
    const int T1 = 2048;                // v4u offset of t=1 K tile (16*DM/8)

    for (int it = 0; it < 32; it++) {
        // --- cooperative V stage: 4 dword loads, 8 b16 LDS stores
#pragma unroll
        for (int i = 0; i < 4; i++) {
            unsigned int dv = *(const unsigned int*)(vrow + i * 8 * DM);
            vsb[i * 8]      = ((const bf16_t*)&dv)[0];
            vsb[LP + i * 8] = ((const bf16_t*)&dv)[1];
        }

        // --- score tiles: S[16q x 32k] = Q x K^T
        v8f s[2];
        {
            Frag k0, k1;
            k0.q[0] = kp[hs];          k0.q[1] = kp[hs + 2];
            k1.q[0] = kp[4 + hs];      k1.q[1] = kp[4 + hs + 2];
            v8f z = v8f_zero();
            z = wmma_bf16(qa[0], k0, z);
            s[0] = wmma_bf16(qa[1], k1, z);
            k0.q[0] = kp[T1 + hs];     k0.q[1] = kp[T1 + hs + 2];
            k1.q[0] = kp[T1 + 4 + hs]; k1.q[1] = kp[T1 + 4 + hs + 2];
            z = v8f_zero();
            z = wmma_bf16(qa[0], k0, z);
            s[1] = wmma_bf16(qa[1], k1, z);
        }

        // --- bias (LDS table) + online softmax (C layout: m=r+8hs, n=lane16)
#pragma unroll
        for (int r = 0; r < 8; r++) {
            s[0][r] += biasp[-r];
            s[1][r] += biasp[16 - r];
        }

#pragma unroll
        for (int r = 0; r < 8; r++) {
            float mx = fmaxf(s[0][r], s[1][r]);
#pragma unroll
            for (int off = 1; off < 16; off <<= 1)
                mx = fmaxf(mx, __shfl_xor(mx, off, 32));
            float newm = fmaxf(mrow[r], mx);
            float scale = __expf(mrow[r] - newm);
            mrow[r] = newm;
            s[0][r] = __expf(s[0][r] - newm);
            s[1][r] = __expf(s[1][r] - newm);
            accl[r] *= scale;
#pragma unroll
            for (int tn = 0; tn < 4; tn++) acc[tn][r] *= scale;
        }

        // --- P (C layout) -> LDS -> A layout, bf16
#pragma unroll
        for (int t = 0; t < 2; t++)
#pragma unroll
            for (int r = 0; r < 8; r++)
                P[(r + 8 * hs) * LP + t * 16 + lane16] = (bf16_t)s[t][r];

        __syncthreads();  // V tile complete + own P stores complete

        Frag pa;
#pragma unroll
        for (int v = 0; v < 8; v++)
            pa.u[v] = *(const unsigned int*)&P[lane16 * LP + frag_kbase(v, hs)];
        Frag vf[4];
#pragma unroll
        for (int tn = 0; tn < 4; tn++)
#pragma unroll
            for (int v = 0; v < 8; v++)
                vf[tn].u[v] = *(const unsigned int*)&Vs[(tn * 16 + lane16) * LP + frag_kbase(v, hs)];
        accl = wmma_bf16(pa, onesf, accl);   // denominator column
#pragma unroll
        for (int tn = 0; tn < 4; tn++)
            acc[tn] = wmma_bf16(pa, vf[tn], acc[tn]);

        __syncthreads();  // all waves done reading Vs before next overwrite

        kp    += KSTEP / 8;   // 32 rows, in v4u units
        vrow  += KSTEP;
        biasp += 32;
    }

    // normalize + write bf16 [B,S,H*HD]
#pragma unroll
    for (int r = 0; r < 8; r++) {
        float inv = 1.0f / accl[r];
        int qp = q0 + r + 8 * hs;
#pragma unroll
        for (int tn = 0; tn < 4; tn++) {
            int d = tn * 16 + lane16;
            O[(size_t)(b * S + qp) * DM + h * HD + d] = (bf16_t)(acc[tn][r] * inv);
        }
    }
}

// ---------------------------------------------------------------------
// Host-side orchestration
// ---------------------------------------------------------------------
extern "C" void kernel_launch(void* const* d_in, const int* in_sizes, int n_in,
                              void* d_out, int out_size, void* d_ws, size_t ws_size,
                              hipStream_t stream) {
    const float* x   = (const float*)d_in[0];
    const float* ln1 = (const float*)d_in[1];
    const float* wq  = (const float*)d_in[2];
    const float* wk  = (const float*)d_in[3];
    const float* wv  = (const float*)d_in[4];
    const float* wo  = (const float*)d_in[5];
    const float* ln2 = (const float*)d_in[6];
    const float* wi  = (const float*)d_in[7];
    const float* wm  = (const float*)d_in[8];
    const float* rel = (const float*)d_in[9];
    float* out = (float*)d_out;

    const int Brows = 8 * 1024;  // B*S = 8192
    char* ws = (char*)d_ws;
    const size_t MB = 1ull << 20;
    bf16_t* hb  = (bf16_t*)(ws + 0);        // 16MB (also reused for MLP input y)
    bf16_t* qb  = (bf16_t*)(ws + 16 * MB);  // 16MB
    bf16_t* kb  = (bf16_t*)(ws + 32 * MB);  // 16MB
    bf16_t* vb  = (bf16_t*)(ws + 48 * MB);  // 16MB
    bf16_t* ab  = (bf16_t*)(ws + 64 * MB);  // 16MB attn out
    bf16_t* wqb = (bf16_t*)(ws + 80 * MB);  // 2MB
    bf16_t* wkb = (bf16_t*)(ws + 82 * MB);
    bf16_t* wvb = (bf16_t*)(ws + 84 * MB);
    bf16_t* wob = (bf16_t*)(ws + 86 * MB);
    bf16_t* wib = (bf16_t*)(ws + 88 * MB);  // 8MB
    bf16_t* wmb = (bf16_t*)(ws + 96 * MB);  // 8MB
    bf16_t* m1b = (bf16_t*)(ws + 104 * MB); // 64MB ReLU activations
    float*  btab = (float*)(ws + 168 * MB); // 128KB bias table

    auto conv = [&](const float* src, bf16_t* dst, int n) {
        f32_to_bf16_kernel<<<(n + 255) / 256, 256, 0, stream>>>(src, dst, n);
    };
    conv(wq, wqb, 1024 * 1024);
    conv(wk, wkb, 1024 * 1024);
    conv(wv, wvb, 1024 * 1024);
    conv(wo, wob, 1024 * 1024);
    conv(wi, wib, 1024 * 4096);
    conv(wm, wmb, 4096 * 1024);

    bias_table_kernel<<<16, 256, 0, stream>>>(rel, btab);

    // h = rmsnorm(x) -> bf16
    rmsnorm_kernel<<<Brows, 256, 0, stream>>>(x, ln1, hb, 1024);

    dim3 gD(1024 / BN, Brows / BM);   // (4, 64)
    dim3 gM(4096 / BN, Brows / BM);   // (16, 64)

    // Q, K, V projections (bf16 out)
    gemm_bf16_kernel<0><<<gD, 256, 0, stream>>>(hb, wqb, qb, nullptr, Brows, 1024, 1024);
    gemm_bf16_kernel<0><<<gD, 256, 0, stream>>>(hb, wkb, kb, nullptr, Brows, 1024, 1024);
    gemm_bf16_kernel<0><<<gD, 256, 0, stream>>>(hb, wvb, vb, nullptr, Brows, 1024, 1024);

    // attention with fused relative-position bias + streaming softmax
    flash_attn_kernel<<<1024, 256, 0, stream>>>(qb, kb, vb, btab, ab);

    // x2 = x + attn @ wo  (fp32, into d_out)
    gemm_bf16_kernel<1><<<gD, 256, 0, stream>>>(ab, wob, out, x, Brows, 1024, 1024);

    // y = rmsnorm(x2) -> bf16 (reuse hb)
    rmsnorm_kernel<<<Brows, 256, 0, stream>>>(out, ln2, hb, 1024);

    // m1 = relu(y @ wi) -> bf16
    gemm_bf16_kernel<2><<<gM, 256, 0, stream>>>(hb, wib, m1b, nullptr, Brows, 4096, 1024);

    // out = x2 + m1 @ wo_mlp  (fp32)
    gemm_bf16_kernel<1><<<gD, 256, 0, stream>>>(m1b, wmb, out, out, Brows, 1024, 4096);
}
